// MyConv2d_30786325578325
// MI455X (gfx1250) — compile-verified, matrix operand order checked by reading, not run on hardware
//
#include <hip/hip_runtime.h>

// ---------------------------------------------------------------------------
// Conv2d 3x3 s1 p1, NCHW fp32 (N=32, C=128, H=W=56, Cout=256)
// Implicit GEMM on CDNA5 WMMA bf16 (16x16x32), bf16x3 split for ~fp32 accuracy.
// Data movement via Tensor Data Mover (TENSOR_LOAD_TO_LDS) + TENSORcnt,
// double-buffered LDS, 8x8 spatial tiles so all TDM tiles have constant stride.
// ---------------------------------------------------------------------------

typedef __attribute__((ext_vector_type(16))) __bf16        v16bf;
typedef __attribute__((ext_vector_type(8)))  float         v8f;
typedef __attribute__((ext_vector_type(4)))  unsigned int  u32x4;
typedef __attribute__((ext_vector_type(8)))  int           i32x8;
typedef __attribute__((ext_vector_type(4)))  int           i32x4;

union Frag {
    int4  q[2];   // two 16B LDS loads
    v16bf v;      // WMMA operand (32B = 8 VGPRs)
};

// ---- constants -------------------------------------------------------------
#define NB      32
#define CIN     128
#define COUT    256
#define HW      56
#define HWP     58          // padded
#define SPAT    (HW*HW)     // 3136
#define SPATP   (HWP*HWP)   // 3364
#define CO_TILE 128

// LDS buffer layout (per phase): A rows 80B * 128, B rows 80B * 64
#define LDS_AH   0
#define LDS_AL   10240
#define LDS_BH   20480
#define LDS_BL   25600
#define LDS_PHASE 30720     // one phase size; x2 double buffered

// workspace byte offsets
#define WH_OFF  0ull
#define WL_OFF  589824ull                       // 9*128*256*2
#define XH_OFF  1179648ull
#define XL_OFF  (XH_OFF + 27549696ull)          // 32*(58*58-1)*128*2
#define WS_NEEDED (XL_OFF + 27549696ull)        // ~56.3 MB

// ---- bf16 helpers (round-to-nearest-even, manual) --------------------------
static __device__ __forceinline__ unsigned short f2bf(float f) {
    unsigned u = __float_as_uint(f);
    unsigned r = u + 0x7FFFu + ((u >> 16) & 1u);
    return (unsigned short)(r >> 16);
}
static __device__ __forceinline__ float bf2f(unsigned short h) {
    return __uint_as_float(((unsigned)h) << 16);
}

// ---------------------------------------------------------------------------
// TDM descriptor build + issue (6-arg clang-23 builtin form).
// 2B elements, rows of 32 elems (64B), LDS pad 16B after every 64B stored
// -> LDS row stride 80B (bank-conflict-free b128 fragment loads).
// 2D tile: tile = 32 x tile1           (stride0 elems between rows)
// 3D tile: tile = 32 x tile1 x tile2   (stride1 elems between planes)
// ---------------------------------------------------------------------------
static __device__ __forceinline__ void tdm_load(unsigned long long gaddr,
                                                unsigned ldsOff,
                                                int tile1, int tile2,
                                                int stride0, int stride1) {
    u32x4 g0;
    i32x8 g1;
    i32x4 g2, g3;
    i32x8 g4;
    g0[0] = 1u;                                              // count=1, user mode
    g0[1] = ldsOff;                                          // lds_addr [63:32]
    g0[2] = (unsigned)gaddr;                                 // global_addr lo
    g0[3] = ((unsigned)(gaddr >> 32) & 0x01FFFFFFu)          // global_addr hi
            | 0x80000000u;                                   // type=2 (image)

    // data_size=2B (1<<16), pad_enable (1<<20), pad_interval=16DW (3<<22),
    // pad_amount=4DW (3<<25), workgroup_mask=0
    g1[0] = 0x06D10000;
    g1[1] = 32 << 16;                                        // tensor_dim0 = 32
    g1[2] = (tile1 & 0xffff) << 16;                          // tensor_dim1 lo
    g1[3] = 32 << 16;                                        // tile_dim0 = 32
    g1[4] = (tile1 & 0xffff) | ((tile2 & 0xffff) << 16);     // tile_dim1, tile_dim2
    g1[5] = stride0;                                         // dim0_stride lo32
    g1[6] = (stride1 & 0xffff) << 16;                        // dim1_stride lo16
    g1[7] = (int)(((unsigned)stride1) >> 16);                // dim1_stride hi

    g2[0] = tile2;                                           // tensor_dim2
    g2[1] = 0; g2[2] = 0; g2[3] = 0;
    g3[0] = 0; g3[1] = 0; g3[2] = 0; g3[3] = 0;
    g4[0] = 0; g4[1] = 0; g4[2] = 0; g4[3] = 0;
    g4[4] = 0; g4[5] = 0; g4[6] = 0; g4[7] = 0;

    __builtin_amdgcn_tensor_load_to_lds(g0, g1, g2, g3, g4, 0);
}

// ---------------------------------------------------------------------------
// Pre-pass 1: weights OIHW fp32 -> bf16 hi/lo, layout [khkw][cblk][co][ci]
// ---------------------------------------------------------------------------
__global__ void wtrans_kernel(const float* __restrict__ w,
                              unsigned short* __restrict__ Wh,
                              unsigned short* __restrict__ Wl) {
    int tid = blockIdx.x * 256 + threadIdx.x;            // 0 .. 294911
    int ci   = tid & 31;
    int co   = (tid >> 5) & 255;
    int cb   = (tid >> 13) & 3;
    int khkw = tid >> 15;
    int c    = cb * 32 + ci;
    float v = w[(co * CIN + c) * 9 + khkw];
    unsigned short h = f2bf(v);
    unsigned short l = f2bf(v - bf2f(h));
    Wh[tid] = h;
    Wl[tid] = l;
}

// ---------------------------------------------------------------------------
// Pre-pass 2: x NCHW fp32 -> zero-padded NHWC bf16 hi/lo [n][58][58][128]
// ---------------------------------------------------------------------------
__global__ void xtrans_kernel(const float* __restrict__ x,
                              unsigned short* __restrict__ Xh,
                              unsigned short* __restrict__ Xl) {
    int tid = blockIdx.x * 256 + threadIdx.x;            // 0 .. 13774847
    int c  = tid & 127;
    int r  = tid >> 7;
    int wp = r % HWP;
    int r2 = r / HWP;
    int hp = r2 % HWP;
    int n  = r2 / HWP;
    float v = 0.f;
    if (hp >= 1 && hp <= HW && wp >= 1 && wp <= HW) {
        v = x[(n * CIN + c) * SPAT + (hp - 1) * HW + (wp - 1)];
    }
    unsigned short h = f2bf(v);
    unsigned short l = f2bf(v - bf2f(h));
    Xh[tid] = h;
    Xl[tid] = l;
}

// ---------------------------------------------------------------------------
// Main kernel: grid (49 spatial 8x8 tiles, 2 co-tiles, 32 images), 256 thr.
// Wave 0 drives TDM loads (double buffered); all 8 waves run WMMA.
// ---------------------------------------------------------------------------
__global__ void __launch_bounds__(256)
conv_wmma_kernel(const unsigned short* __restrict__ Xh,
                 const unsigned short* __restrict__ Xl,
                 const unsigned short* __restrict__ Wh,
                 const unsigned short* __restrict__ Wl,
                 const float* __restrict__ bias,
                 float* __restrict__ out) {
    __shared__ __align__(16) unsigned char smem[2 * LDS_PHASE];

    const int t    = threadIdx.x;
    const int lane = t & 31;
    const int wave = t >> 5;
    const int wm   = wave & 1;            // co half (64 rows)
    const int wn   = wave >> 1;           // spatial quarter (16 positions)

    const int tileIdx = blockIdx.x;       // 0..48  -> 7x7 grid of 8x8 tiles
    const int ty = tileIdx / 7;
    const int tx = tileIdx - ty * 7;
    const int hBase = ty * 8;
    const int wBase = tx * 8;
    const int coBase = blockIdx.y * CO_TILE;
    const int n      = blockIdx.z;
    const int xNB    = n * SPATP;

    const unsigned ldsBase = (unsigned)(size_t)(void*)smem;
    const unsigned long long whA = (unsigned long long)(size_t)Wh;
    const unsigned long long wlA = (unsigned long long)(size_t)Wl;
    const unsigned long long xhA = (unsigned long long)(size_t)Xh;
    const unsigned long long xlA = (unsigned long long)(size_t)Xl;

    // fragment-load coords
    const int half = lane >> 4;           // K half-group
    const int l16  = lane & 15;
    const int off1 = half * 16;
    const int off2 = 32 + half * 16;
    const int bRowOff = (wn * 16 + l16) * 80;

    v8f acc[4];
    #pragma unroll
    for (int m = 0; m < 4; ++m)
        #pragma unroll
        for (int r = 0; r < 8; ++r) acc[m][r] = 0.f;

    // issue one k-step's 4 tensor loads (uniform; wave0 only)
    auto issue_step = [&](int s, int buf) {
        const unsigned base = ldsBase + buf * LDS_PHASE;
        // A: weights, contiguous 128 rows x 64B
        unsigned long long aOff =
            ((unsigned long long)(s * COUT + coBase)) * 32ull * 2ull;
        tdm_load(whA + aOff, base + LDS_AH, 128, 0, 32, 0);
        tdm_load(wlA + aOff, base + LDS_AL, 128, 0, 32, 0);
        // B: x tile, 32c x 8w x 8h from padded NHWC
        const int khkw = s >> 2, cb = s & 3;
        const int kh = khkw / 3, kw = khkw - kh * 3;
        unsigned long long bOff =
            ((unsigned long long)((xNB + (hBase + kh) * HWP + (wBase + kw)) * CIN
                                  + cb * 32)) * 2ull;
        tdm_load(xhA + bOff, base + LDS_BH, 8, 8, CIN, HWP * CIN);
        tdm_load(xlA + bOff, base + LDS_BL, 8, 8, CIN, HWP * CIN);
    };

    if (wave == 0) issue_step(0, 0);

    #pragma unroll 1
    for (int s = 0; s < 36; ++s) {
        const int buf = s & 1;
        if (wave == 0) {
            if (s + 1 < 36) {
                issue_step(s + 1, buf ^ 1);
                __builtin_amdgcn_s_wait_tensorcnt(4);  // in-order: step s done
            } else {
                __builtin_amdgcn_s_wait_tensorcnt(0);
            }
        }
        __syncthreads();

        const unsigned char* pb = smem + buf * LDS_PHASE;
        Frag bh, bl;
        bh.q[0] = *reinterpret_cast<const int4*>(pb + LDS_BH + bRowOff + off1);
        bh.q[1] = *reinterpret_cast<const int4*>(pb + LDS_BH + bRowOff + off2);
        bl.q[0] = *reinterpret_cast<const int4*>(pb + LDS_BL + bRowOff + off1);
        bl.q[1] = *reinterpret_cast<const int4*>(pb + LDS_BL + bRowOff + off2);

        #pragma unroll
        for (int m = 0; m < 4; ++m) {
            const int aRowOff = (wm * 64 + m * 16 + l16) * 80;
            Frag ah, al;
            ah.q[0] = *reinterpret_cast<const int4*>(pb + LDS_AH + aRowOff + off1);
            ah.q[1] = *reinterpret_cast<const int4*>(pb + LDS_AH + aRowOff + off2);
            al.q[0] = *reinterpret_cast<const int4*>(pb + LDS_AL + aRowOff + off1);
            al.q[1] = *reinterpret_cast<const int4*>(pb + LDS_AL + aRowOff + off2);

            acc[m] = __builtin_amdgcn_wmma_f32_16x16x32_bf16(
                false, ah.v, false, bh.v, (short)0, acc[m], false, false);
            acc[m] = __builtin_amdgcn_wmma_f32_16x16x32_bf16(
                false, ah.v, false, bl.v, (short)0, acc[m], false, false);
            acc[m] = __builtin_amdgcn_wmma_f32_16x16x32_bf16(
                false, al.v, false, bh.v, (short)0, acc[m], false, false);
        }
        __syncthreads();
    }

    // ---- epilogue: + bias, store NCHW ----
    // LDS B row r = h_local*8 + w_local ; subtile wn covers r = wn*16 + l16
    const int r0 = wn * 16 + l16;
    const int pOut = (hBase + (r0 >> 3)) * HW + wBase + (r0 & 7);
    #pragma unroll
    for (int m = 0; m < 4; ++m) {
        const int co0 = coBase + wm * 64 + m * 16 + (half << 3);
        #pragma unroll
        for (int r = 0; r < 8; ++r) {
            const int co = co0 + r;
            out[(n * COUT + co) * SPAT + pOut] = acc[m][r] + bias[co];
        }
    }
}

// ---------------------------------------------------------------------------
// Fallback: naive direct conv (used only if workspace is too small)
// ---------------------------------------------------------------------------
__global__ void naive_conv_kernel(const float* __restrict__ x,
                                  const float* __restrict__ w,
                                  const float* __restrict__ bias,
                                  float* __restrict__ out, int total) {
    int tid = blockIdx.x * 256 + threadIdx.x;
    if (tid >= total) return;
    int pp = tid % SPAT;
    int co = (tid / SPAT) % COUT;
    int n  = tid / (SPAT * COUT);
    int h = pp / HW, ww = pp % HW;
    float acc = bias[co];
    for (int c = 0; c < CIN; ++c)
        for (int kh = 0; kh < 3; ++kh) {
            int ih = h + kh - 1;
            if (ih < 0 || ih >= HW) continue;
            for (int kw = 0; kw < 3; ++kw) {
                int iw = ww + kw - 1;
                if (iw < 0 || iw >= HW) continue;
                acc += x[(n * CIN + c) * SPAT + ih * HW + iw] *
                       w[(co * CIN + c) * 9 + kh * 3 + kw];
            }
        }
    out[tid] = acc;
}

// ---------------------------------------------------------------------------
extern "C" void kernel_launch(void* const* d_in, const int* in_sizes, int n_in,
                              void* d_out, int out_size, void* d_ws, size_t ws_size,
                              hipStream_t stream) {
    const float* x    = (const float*)d_in[0];
    const float* w    = (const float*)d_in[1];
    const float* bias = (const float*)d_in[2];
    float* out        = (float*)d_out;

    if (ws_size < WS_NEEDED) {
        int total = NB * COUT * SPAT;
        naive_conv_kernel<<<(total + 255) / 256, 256, 0, stream>>>(x, w, bias, out, total);
        return;
    }

    unsigned short* Wh = (unsigned short*)((char*)d_ws + WH_OFF);
    unsigned short* Wl = (unsigned short*)((char*)d_ws + WL_OFF);
    unsigned short* Xh = (unsigned short*)((char*)d_ws + XH_OFF);
    unsigned short* Xl = (unsigned short*)((char*)d_ws + XL_OFF);

    wtrans_kernel<<<1152, 256, 0, stream>>>(w, Wh, Wl);
    xtrans_kernel<<<53808, 256, 0, stream>>>(x, Xh, Xl);

    dim3 grid(49, COUT / CO_TILE, NB);   // (49, 2, 32)
    conv_wmma_kernel<<<grid, 256, 0, stream>>>(Xh, Xl, Wh, Wl, bias, out);
}
